// SRNN_55671366091071
// MI455X (gfx1250) — matrix operand order repaired and checked by reading, not compiled, not run on hardware
//
#include <hip/hip_runtime.h>
#include <hip/hip_bf16.h>
#include <math.h>

// Problem sizes (fixed by the reference)
constexpr int Bc = 64, Sc = 512, Vc = 1024, Hc = 2048;

typedef __bf16 bf16_t;
typedef __attribute__((ext_vector_type(16))) __bf16 bf16x16;  // 8 VGPRs
typedef __attribute__((ext_vector_type(8)))  float   f32x8;   // 8 VGPRs

union FragU { bf16x16 v; uint4 q[2]; };

#define WMMA_BF16(a, b, c) \
    __builtin_amdgcn_wmma_f32_16x16x32_bf16(false, (a), false, (b), (short)0, (c), false, false)

// ---------------------------------------------------------------------------
// Fragment loaders (CDNA5 wave32 WMMA layouts, 16-bit A 16x32 / B 32x16)
// ---------------------------------------------------------------------------
__device__ __forceinline__ bf16x16 load_frag_a(const bf16_t* __restrict__ base,
                                               size_t lda, int k0, int lane) {
    const int m  = lane & 15;
    const int hh = lane >> 4;
    const bf16_t* p = base + (size_t)m * lda + (size_t)(k0 + hh * 8);
    FragU f;
    f.q[0] = *reinterpret_cast<const uint4*>(p);        // K = hh*8 + 0..7   (VGPR0-3)
    f.q[1] = *reinterpret_cast<const uint4*>(p + 16);   // K = 16+hh*8+0..7  (VGPR4-7)
    return f.v;
}

// A fragment from an LDS-resident [64][32] bf16 K-slice (rows = batch, cols = K chunk)
__device__ __forceinline__ bf16x16 load_frag_a_lds(const bf16_t* sA, int mt, int lane) {
    const int m  = lane & 15;
    const int hh = lane >> 4;
    const bf16_t* p = sA + (size_t)(mt * 16 + m) * 32 + hh * 8;
    FragU f;
    f.q[0] = *reinterpret_cast<const uint4*>(p);        // ds_load_b128
    f.q[1] = *reinterpret_cast<const uint4*>(p + 16);
    return f.v;
}

__device__ __forceinline__ bf16x16 load_frag_b(const bf16_t* __restrict__ base,
                                               size_t ldb, int k0, int lane) {
    const int n  = lane & 15;
    const int hh = lane >> 4;
    const bf16_t* p = base + (size_t)n * ldb + (size_t)(k0 + hh * 8);
    FragU f;
    f.q[0] = *reinterpret_cast<const uint4*>(p);        // K = hh*8 + 0..7   (VGPR0-3)
    f.q[1] = *reinterpret_cast<const uint4*>(p + 16);   // K = 16+hh*8+0..7  (VGPR4-7)
    return f.v;
}

// ---------------------------------------------------------------------------
// Async global->LDS staging of one 64x32 bf16 K-slice of h (4 KB per chunk).
// 256 threads x 16 B = 4 KB in one async-to-LDS instruction per wave.
// ---------------------------------------------------------------------------
__device__ __forceinline__ void async_stage_h(const bf16_t* __restrict__ hbf, int k0,
                                              uint32_t lds_off, int tid) {
    const int r  = tid >> 2;            // row 0..63
    const int c0 = (tid & 3) * 8;       // K sub-offset 0,8,16,24
    const bf16_t* gp = hbf + (size_t)r * Hc + (size_t)(k0 + c0);
    // VDST = LDS byte offset, ADDR = 64-bit global address (GV mode); ASYNCcnt-tracked.
    asm volatile("global_load_async_to_lds_b128 %0, %1, off"
                 :: "v"(lds_off), "v"(gp)
                 : "memory");
}

__device__ __forceinline__ void wait_async_le1() {
    asm volatile("s_wait_asynccnt 1" ::: "memory");
}
__device__ __forceinline__ void wait_async_le0() {
    asm volatile("s_wait_asynccnt 0" ::: "memory");
}

// ---------------------------------------------------------------------------
// Elementwise prep kernels
// ---------------------------------------------------------------------------
__global__ void k_f32_to_bf16(const float* __restrict__ src, bf16_t* __restrict__ dst, size_t n) {
    size_t i = (size_t)blockIdx.x * blockDim.x + threadIdx.x;
    size_t stride = (size_t)gridDim.x * blockDim.x;
    for (; i < n; i += stride) dst[i] = (bf16_t)src[i];
}

__global__ void k_init_h(const float* __restrict__ h0, bf16_t* __restrict__ hbf) {
    int i = blockIdx.x * blockDim.x + threadIdx.x;      // Bc*Hc threads
    hbf[i] = (bf16_t)h0[i & (Hc - 1)];                  // broadcast row h0[1,H] -> [B,H]
}

// ---------------------------------------------------------------------------
// GEMM 1: xp[s][b][h] = sum_v x[b][s][v] * Wx[h][v]
// Wave computes one (s, mt) x 4 consecutive n-tiles: A fragment reused 4x.
// groups = Sc * 4 * 32 = 65536 waves -> 8192 blocks of 256.
// ---------------------------------------------------------------------------
__global__ void k_xp_gemm(const bf16_t* __restrict__ xbf, const bf16_t* __restrict__ Wx,
                          float* __restrict__ xp) {
    const int lane = threadIdx.x & 31;
    const int wid  = threadIdx.x >> 5;
    const int NG = (Hc / 16) / 4;            // 32 groups of 4 n-tiles
    const int MT = Bc / 16;                  // 4
    int grp = blockIdx.x * 8 + wid;
    int s  = grp / (MT * NG);
    int r  = grp % (MT * NG);
    int mt = r / NG;
    int ng = r % NG;

    const bf16_t* Abase = xbf + ((size_t)(mt * 16) * Sc + s) * Vc;  // row stride Sc*Vc
    const bf16_t* Bbase = Wx + (size_t)(ng * 64) * Vc;

    f32x8 c[4];
    #pragma unroll
    for (int t = 0; t < 4; ++t) c[t] = f32x8{0.f, 0.f, 0.f, 0.f, 0.f, 0.f, 0.f, 0.f};

    for (int k0 = 0; k0 < Vc; k0 += 32) {
        __builtin_prefetch(Abase + (size_t)(lane & 15) * Sc * Vc + k0 + 64, 0, 1);
        bf16x16 a = load_frag_a(Abase, (size_t)Sc * Vc, k0, lane);
        #pragma unroll
        for (int t = 0; t < 4; ++t) {
            bf16x16 b = load_frag_b(Bbase + (size_t)(t * 16) * Vc, Vc, k0, lane);
            c[t] = WMMA_BF16(a, b, c[t]);
        }
    }

    const int n  = lane & 15;
    const int hh = lane >> 4;
    #pragma unroll
    for (int t = 0; t < 4; ++t) {
        float* out = xp + (size_t)s * (Bc * Hc) + (size_t)(mt * 16) * Hc
                     + (size_t)(ng * 64 + t * 16) + n;
        #pragma unroll
        for (int j = 0; j < 8; ++j) {
            int m = j + hh * 8;
            out[(size_t)m * Hc] = c[t][j];
        }
    }
}

// ---------------------------------------------------------------------------
// Per-step GEMM Y: logits[b][v] = sum_h hbf[b][h] * Wy[v][h] + by[v]
// Wave owns one n-tile x all 4 m-tiles (B fragment loaded once -> Wy read 1x).
// h K-slices staged into LDS via async-to-LDS, double buffered.
// 64 n-tiles -> 8 blocks of 256 (8 waves).
// ---------------------------------------------------------------------------
__global__ void k_y_gemm(const bf16_t* __restrict__ hbf, const bf16_t* __restrict__ Wy,
                         const float* __restrict__ by, float* __restrict__ logits) {
    __shared__ __align__(16) bf16_t sA[2][64][32];     // 8 KB, double-buffered K-slices
    const int tid  = threadIdx.x;
    const int lane = tid & 31;
    const int wid  = tid >> 5;
    const int nt   = blockIdx.x * 8 + wid;             // 0..63

    const bf16_t* Bbase = Wy + (size_t)(nt * 16) * Hc;
    const uint32_t lds0 = (uint32_t)(uintptr_t)(&sA[0][0][0]) + (uint32_t)tid * 16u;

    f32x8 c[4];
    #pragma unroll
    for (int mt = 0; mt < 4; ++mt) c[mt] = f32x8{0.f, 0.f, 0.f, 0.f, 0.f, 0.f, 0.f, 0.f};

    const int NK = Hc / 32;                            // 64 K-chunks
    async_stage_h(hbf, 0, lds0, tid);                  // prime buffer 0
    for (int i = 0; i < NK; ++i) {
        const int cur = i & 1;
        if (i + 1 < NK) {
            async_stage_h(hbf, (i + 1) * 32, lds0 + (uint32_t)((1 - cur) * 4096), tid);
            wait_async_le1();                          // buffer `cur` complete (in-order)
        } else {
            wait_async_le0();
        }
        __syncthreads();                               // all waves' stages visible

        bf16x16 b = load_frag_b(Bbase, Hc, i * 32, lane);
        #pragma unroll
        for (int mt = 0; mt < 4; ++mt) {
            bf16x16 a = load_frag_a_lds(&sA[cur][0][0], mt, lane);
            c[mt] = WMMA_BF16(a, b, c[mt]);
        }
        __syncthreads();                               // reads done before refill of `cur`
    }

    const int n  = lane & 15;
    const int hh = lane >> 4;
    const int col = nt * 16 + n;
    const float bias = by[col];
    #pragma unroll
    for (int mt = 0; mt < 4; ++mt) {
        #pragma unroll
        for (int j = 0; j < 8; ++j) {
            int row = mt * 16 + j + hh * 8;
            logits[(size_t)row * Vc + col] = c[mt][j] + bias;
        }
    }
}

// ---------------------------------------------------------------------------
// Per-step GEMM H: h'[b][h] = tanh(xp_t[b][h] + sum_k hbf[b][k]*Wh[h][k] + bh[h])
// Same scheme: one n-tile x 4 m-tiles per wave, LDS-staged h slices.
// 128 n-tiles -> 16 blocks of 256.
// ---------------------------------------------------------------------------
__global__ void k_h_gemm(const bf16_t* __restrict__ hbf, const bf16_t* __restrict__ Wh,
                         const float* __restrict__ xp_t, const float* __restrict__ bh,
                         bf16_t* __restrict__ hnext, float* __restrict__ hfin) {
    __shared__ __align__(16) bf16_t sA[2][64][32];
    const int tid  = threadIdx.x;
    const int lane = tid & 31;
    const int wid  = tid >> 5;
    const int nt   = blockIdx.x * 8 + wid;             // 0..127

    const bf16_t* Bbase = Wh + (size_t)(nt * 16) * Hc;
    const uint32_t lds0 = (uint32_t)(uintptr_t)(&sA[0][0][0]) + (uint32_t)tid * 16u;

    f32x8 c[4];
    #pragma unroll
    for (int mt = 0; mt < 4; ++mt) c[mt] = f32x8{0.f, 0.f, 0.f, 0.f, 0.f, 0.f, 0.f, 0.f};

    const int NK = Hc / 32;
    async_stage_h(hbf, 0, lds0, tid);
    for (int i = 0; i < NK; ++i) {
        const int cur = i & 1;
        if (i + 1 < NK) {
            async_stage_h(hbf, (i + 1) * 32, lds0 + (uint32_t)((1 - cur) * 4096), tid);
            wait_async_le1();
        } else {
            wait_async_le0();
        }
        __syncthreads();

        bf16x16 b = load_frag_b(Bbase, Hc, i * 32, lane);
        #pragma unroll
        for (int mt = 0; mt < 4; ++mt) {
            bf16x16 a = load_frag_a_lds(&sA[cur][0][0], mt, lane);
            c[mt] = WMMA_BF16(a, b, c[mt]);
        }
        __syncthreads();
    }

    const int n  = lane & 15;
    const int hh = lane >> 4;
    const int col = nt * 16 + n;
    const float bias = bh[col];
    #pragma unroll
    for (int mt = 0; mt < 4; ++mt) {
        #pragma unroll
        for (int j = 0; j < 8; ++j) {
            int row = mt * 16 + j + hh * 8;
            float val = tanhf(xp_t[(size_t)row * Hc + col] + c[mt][j] + bias);
            hnext[(size_t)row * Hc + col] = (bf16_t)val;
            if (hfin) hfin[(size_t)row * Hc + col] = val;
        }
    }
}

// ---------------------------------------------------------------------------
// Softmax over V=1024 per row; writes y[b][t][:]. One block (256 thr) per row.
// ---------------------------------------------------------------------------
__global__ void k_softmax(const float* __restrict__ logits, float* __restrict__ y, int t) {
    const int b   = blockIdx.x;
    const int tid = threadIdx.x;
    __shared__ float red[256];

    const float* row = logits + (size_t)b * Vc;
    float v[4];
    float vmax = -INFINITY;
    #pragma unroll
    for (int i = 0; i < 4; ++i) {
        v[i] = row[tid + i * 256];
        vmax = fmaxf(vmax, v[i]);
    }
    red[tid] = vmax;
    __syncthreads();
    for (int s2 = 128; s2 > 0; s2 >>= 1) {
        if (tid < s2) red[tid] = fmaxf(red[tid], red[tid + s2]);
        __syncthreads();
    }
    vmax = red[0];
    __syncthreads();

    float sum = 0.f;
    #pragma unroll
    for (int i = 0; i < 4; ++i) {
        v[i] = __expf(v[i] - vmax);
        sum += v[i];
    }
    red[tid] = sum;
    __syncthreads();
    for (int s2 = 128; s2 > 0; s2 >>= 1) {
        if (tid < s2) red[tid] += red[tid + s2];
        __syncthreads();
    }
    const float inv = 1.f / red[0];

    float* out = y + ((size_t)b * Sc + t) * Vc;
    #pragma unroll
    for (int i = 0; i < 4; ++i) out[tid + i * 256] = v[i] * inv;
}

// ---------------------------------------------------------------------------
// Host driver
// ---------------------------------------------------------------------------
extern "C" void kernel_launch(void* const* d_in, const int* in_sizes, int n_in,
                              void* d_out, int out_size, void* d_ws, size_t ws_size,
                              hipStream_t stream) {
    const float* x  = (const float*)d_in[0];   // [B,S,V]
    const float* h0 = (const float*)d_in[1];   // [1,H]
    const float* Wx = (const float*)d_in[2];   // [H,V]
    const float* Wh = (const float*)d_in[3];   // [H,H]
    const float* bh = (const float*)d_in[4];   // [H]
    const float* Wy = (const float*)d_in[5];   // [V,H]
    const float* by = (const float*)d_in[6];   // [V]

    float* y    = (float*)d_out;                         // [B,S,V]
    float* hfin = y + (size_t)Bc * Sc * Vc;              // [B,H]

    // Workspace layout (all offsets 16B-aligned)
    char* w = (char*)d_ws;
    size_t off = 0;
    bf16_t* xbf  = (bf16_t*)(w + off); off += (size_t)Bc * Sc * Vc * 2;   // 67 MB
    bf16_t* Wxb  = (bf16_t*)(w + off); off += (size_t)Hc * Vc * 2;        //  4 MB
    bf16_t* Whb  = (bf16_t*)(w + off); off += (size_t)Hc * Hc * 2;        //  8 MB
    bf16_t* Wyb  = (bf16_t*)(w + off); off += (size_t)Vc * Hc * 2;        //  4 MB
    bf16_t* hbf0 = (bf16_t*)(w + off); off += (size_t)Bc * Hc * 2;
    bf16_t* hbf1 = (bf16_t*)(w + off); off += (size_t)Bc * Hc * 2;
    float*  lg   = (float*)(w + off);  off += (size_t)Bc * Vc * 4;
    float*  xp   = (float*)(w + off);  off += (size_t)Sc * Bc * Hc * 4;   // 268 MB

    // 1) Precision conversion (bf16 inputs for WMMA)
    k_f32_to_bf16<<<4096, 256, 0, stream>>>(x,  xbf, (size_t)Bc * Sc * Vc);
    k_f32_to_bf16<<<2048, 256, 0, stream>>>(Wx, Wxb, (size_t)Hc * Vc);
    k_f32_to_bf16<<<2048, 256, 0, stream>>>(Wh, Whb, (size_t)Hc * Hc);
    k_f32_to_bf16<<<2048, 256, 0, stream>>>(Wy, Wyb, (size_t)Vc * Hc);
    k_init_h<<<(Bc * Hc) / 256, 256, 0, stream>>>(h0, hbf0);

    // 2) Big parallel GEMM: xp = einsum('sbv,hv->sbh')
    {
        int groups = Sc * (Bc / 16) * ((Hc / 16) / 4);   // 65536 waves
        k_xp_gemm<<<groups / 8, 256, 0, stream>>>(xbf, Wxb, xp);
    }

    // 3) Sequential recurrence (graph-capturable fixed launch sequence)
    for (int t = 0; t < Sc; ++t) {
        bf16_t* hc = (t & 1) ? hbf1 : hbf0;
        bf16_t* hn = (t & 1) ? hbf0 : hbf1;
        k_y_gemm<<<(Vc / 16) / 8, 256, 0, stream>>>(hc, Wyb, by, lg);
        k_h_gemm<<<(Hc / 16) / 8, 256, 0, stream>>>(
            hc, Whb, xp + (size_t)t * Bc * Hc, bh, hn,
            (t == Sc - 1) ? hfin : nullptr);
        k_softmax<<<Bc, 256, 0, stream>>>(lg, y, t);
    }
}
